// Attention_16020228014210
// MI455X (gfx1250) — matrix operand order repaired and checked by reading, not compiled, not run on hardware
//
#include <hip/hip_runtime.h>

typedef __attribute__((ext_vector_type(16))) _Float16 v16h;
typedef __attribute__((ext_vector_type(8)))  float    v8f;

#define ATT_DIM 512
#define HID_DIM 1024
#define EMB_DIM 1024
#define SEQ_LEN 2048
#define BATCH   32

__device__ __forceinline__ float tanh_fast(float x) {
    // tanh(x) = 1 - 2/(exp(2x)+1); branch-free, correct limits at +/-inf.
    float e = __expf(2.0f * x);
    return 1.0f - 2.0f * __builtin_amdgcn_rcpf(e + 1.0f);
}

// ---------------------------------------------------------------------------
// Kernel 1: hb[b][a] = hid[b,:] @ W_hid[:,a] + b_hid[a] + b_emb[a]   (32 x 512)
// ---------------------------------------------------------------------------
__global__ void k_hidproj(const float* __restrict__ hid,
                          const float* __restrict__ W,
                          const float* __restrict__ b_hid,
                          const float* __restrict__ b_emb,
                          float* __restrict__ out) {
    int t = blockIdx.x * 256 + threadIdx.x;        // 16384 threads
    int b = t >> 9;
    int a = t & 511;
    float acc = b_hid[a] + b_emb[a];
    const float* h = hid + b * HID_DIM;
    #pragma unroll 4
    for (int k = 0; k < HID_DIM; ++k)
        acc += h[k] * W[k * ATT_DIM + a];
    out[t] = acc;
}

// ---------------------------------------------------------------------------
// Kernel 2: pack W_emb (1024x512 f32) into f16 WMMA B-fragments:
//   lane<16 : column N=lane,    halfs i=0..7 hold K=i,   i=8..15 hold K=16+(i-8)
//   lane>=16: column N=lane-16, halfs i=0..7 hold K=8+i, i=8..15 hold K=24+(i-8)
// Fragment index: [kt(32)][nt(32)][lane(32)][16 halfs]
// ---------------------------------------------------------------------------
__global__ void k_wpack(const float* __restrict__ W, _Float16* __restrict__ Wpk) {
    int t = blockIdx.x * 256 + threadIdx.x;        // 524288 threads
    int i    =  t        & 15;
    int lane = (t >> 4)  & 31;
    int nt   = (t >> 9)  & 31;
    int kt   =  t >> 14;
    int k = kt * 32 + ((i >> 3) << 4) + ((lane >> 4) << 3) + (i & 7);
    int a = nt * 16 + (lane & 15);
    Wpk[t] = (_Float16)W[k * ATT_DIM + a];
}

// ---------------------------------------------------------------------------
// Kernel 3 (main): fused GEMM + tanh + att_v-dot -> scores(B,S).
// 128 threads / 4 waves per WG, 32 rows per WG; A staged once in LDS (64 KB).
// kt loop: 4-deep register pipeline on B (global/L2), 2-deep on A (LDS),
// unconditional clamped prefetch (no branches in the steady state).
// ---------------------------------------------------------------------------
__global__ __launch_bounds__(128) void k_attscore(
    const float*    __restrict__ emb,
    const _Float16* __restrict__ Wpk,
    const float*    __restrict__ hb,      // hid_att + b_hid + b_emb, (32,512)
    const float*    __restrict__ att_v,
    float*          __restrict__ scores)
{
    __shared__ __align__(16) _Float16 Atile[32 * 1024];   // 64 KB

    const int tid  = threadIdx.x;
    const int lane = tid & 31;
    const int wave = tid >> 5;
    const int row0 = blockIdx.x * 32;
    const int n_lo = lane & 15;
    const int hi   = lane >> 4;

    // --- Stage A tile: 32 rows x 1024 cols, f32 -> f16, coalesced float4 ---
    for (int idx = tid; idx < 32 * 256; idx += 128) {
        int m = idx >> 8;
        int c = (idx & 255) << 2;
        float4 f = *(const float4*)(emb + (size_t)(row0 + m) * EMB_DIM + c);
        union { _Float16 h[4]; uint2 u; } cv;
        cv.h[0] = (_Float16)f.x; cv.h[1] = (_Float16)f.y;
        cv.h[2] = (_Float16)f.z; cv.h[3] = (_Float16)f.w;
        *(uint2*)(Atile + m * 1024 + c) = cv.u;
    }
    __syncthreads();

    float scr[2][8];
    #pragma unroll
    for (int t = 0; t < 2; ++t)
        #pragma unroll
        for (int j = 0; j < 8; ++j) scr[t][j] = 0.f;

    union Frag { v16h v; int4 qv[2]; };
    // LDS A-fragment base for this lane (ISA 16-bit A 16x32 layout):
    //   lane<16: halfs K{0..7},{16..23}; lane>=16: K{8..15},{24..31}
    const int abase = n_lo * 1024 + (hi << 3);

    #pragma unroll 1
    for (int q = 0; q < 8; ++q) {
        const int nt = q * 4 + wave;
        v8f acc0 = {0.f,0.f,0.f,0.f,0.f,0.f,0.f,0.f};
        v8f acc1 = {0.f,0.f,0.f,0.f,0.f,0.f,0.f,0.f};

        // B fragment for (kt, nt): 32 B/lane, stride over kt = 16384 halfs
        const _Float16* bbase = Wpk + ((size_t)(nt << 5) + lane) * 16;

        Frag bbuf[4];
        #pragma unroll
        for (int i = 0; i < 4; ++i) {
            const int4* bp = (const int4*)(bbase + (size_t)i * 16384);
            bbuf[i].qv[0] = bp[0];
            bbuf[i].qv[1] = bp[1];
        }
        Frag a0b[2], a1b[2];
        a0b[0].qv[0] = *(const int4*)(Atile + abase);
        a0b[0].qv[1] = *(const int4*)(Atile + abase + 16);
        a1b[0].qv[0] = *(const int4*)(Atile + 16 * 1024 + abase);
        a1b[0].qv[1] = *(const int4*)(Atile + 16 * 1024 + abase + 16);

        #pragma unroll 4
        for (int kt = 0; kt < 32; ++kt) {
            // Prefetch A fragments for kt+1 from LDS (clamped, branch-free)
            const int kn = (kt + 1 < 32) ? (kt + 1) : 31;
            const int an = abase + (kn << 5);
            a0b[(kt + 1) & 1].qv[0] = *(const int4*)(Atile + an);
            a0b[(kt + 1) & 1].qv[1] = *(const int4*)(Atile + an + 16);
            a1b[(kt + 1) & 1].qv[0] = *(const int4*)(Atile + 16 * 1024 + an);
            a1b[(kt + 1) & 1].qv[1] = *(const int4*)(Atile + 16 * 1024 + an + 16);

            acc0 = __builtin_amdgcn_wmma_f32_16x16x32_f16(
                       false, a0b[kt & 1].v, false, bbuf[kt & 3].v,
                       (short)0, acc0, false, false);
            acc1 = __builtin_amdgcn_wmma_f32_16x16x32_f16(
                       false, a1b[kt & 1].v, false, bbuf[kt & 3].v,
                       (short)0, acc1, false, false);

            // Prefetch B fragment for kt+4 into the slot just consumed
            const int kb = (kt + 4 < 32) ? (kt + 4) : 31;
            const int4* bn = (const int4*)(bbase + (size_t)kb * 16384);
            bbuf[kt & 3].qv[0] = bn[0];
            bbuf[kt & 3].qv[1] = bn[1];
        }

        // --- Epilogue: C layout VGPR j -> M=j (+8 if hi), N=lane&15 ---
        const int   a  = (nt << 4) + n_lo;
        const float av = att_v[a];
        #pragma unroll
        for (int t = 0; t < 2; ++t) {
            #pragma unroll
            for (int j = 0; j < 8; ++j) {
                const int m  = t * 16 + hi * 8 + j;
                const int bb = (row0 + m) & 31;
                scr[t][j] += tanh_fast((t == 0 ? acc0[j] : acc1[j])
                                       + hb[bb * ATT_DIM + a]) * av;
            }
        }
    }

    // --- One butterfly reduction over the 16 lanes (columns) per row ---
    __syncthreads();                                 // all waves done with Atile
    float* sPart = (float*)Atile;                    // reuse LDS: [4 waves][32 rows]
    #pragma unroll
    for (int t = 0; t < 2; ++t) {
        #pragma unroll
        for (int j = 0; j < 8; ++j) {
            float v = scr[t][j];
            v += __shfl_xor(v, 1);
            v += __shfl_xor(v, 2);
            v += __shfl_xor(v, 4);
            v += __shfl_xor(v, 8);
            if (n_lo == 0) {
                int m = t * 16 + hi * 8 + j;
                sPart[wave * 32 + m] = v;
            }
        }
    }
    __syncthreads();
    if (tid < 32) {
        float sc = sPart[tid] + sPart[32 + tid] + sPart[64 + tid] + sPart[96 + tid];
        int r = row0 + tid;                          // r = s*32 + b
        scores[(r & 31) * SEQ_LEN + (r >> 5)] = sc;  // scores is (B, S)
    }
}

// ---------------------------------------------------------------------------
// Kernel 4: row softmax over S=2048, one block per batch row.
// ---------------------------------------------------------------------------
__global__ void k_softmax(const float* __restrict__ scores, float* __restrict__ weights) {
    __shared__ float red[256];
    const int b = blockIdx.x, tid = threadIdx.x;
    const float* row = scores + b * SEQ_LEN;
    float v[8];
    float lmax = -1e30f;
    #pragma unroll
    for (int i = 0; i < 8; ++i) { v[i] = row[tid + i * 256]; lmax = fmaxf(lmax, v[i]); }
    red[tid] = lmax; __syncthreads();
    for (int s = 128; s > 0; s >>= 1) {
        if (tid < s) red[tid] = fmaxf(red[tid], red[tid + s]);
        __syncthreads();
    }
    const float m = red[0]; __syncthreads();
    float lsum = 0.f;
    #pragma unroll
    for (int i = 0; i < 8; ++i) { v[i] = expf(v[i] - m); lsum += v[i]; }
    red[tid] = lsum; __syncthreads();
    for (int s = 128; s > 0; s >>= 1) {
        if (tid < s) red[tid] += red[tid + s];
        __syncthreads();
    }
    const float inv = 1.f / red[0];
    #pragma unroll
    for (int i = 0; i < 8; ++i) weights[b * SEQ_LEN + tid + i * 256] = v[i] * inv;
}

// ---------------------------------------------------------------------------
// Kernel 5: context[b][e] = sum_s weights[b][s] * emb[s][b][e]
// ---------------------------------------------------------------------------
__global__ void k_context(const float* __restrict__ emb,
                          const float* __restrict__ weights,
                          float* __restrict__ ctx) {
    const int b  = blockIdx.x;
    const int e  = blockIdx.y * 256 + threadIdx.x;
    const int s0 = blockIdx.z * 256;
    const float* w = weights + b * SEQ_LEN;
    float acc = 0.f;
    for (int s = s0; s < s0 + 256; ++s)
        acc += w[s] * emb[((size_t)s * BATCH + b) * EMB_DIM + e];
    atomicAdd(&ctx[b * EMB_DIM + e], acc);
}

// ---------------------------------------------------------------------------
extern "C" void kernel_launch(void* const* d_in, const int* in_sizes, int n_in,
                              void* d_out, int out_size, void* d_ws, size_t ws_size,
                              hipStream_t stream) {
    const float* hid   = (const float*)d_in[0];   // (32, 1024)
    const float* emb   = (const float*)d_in[1];   // (2048, 32, 1024)
    const float* W_hid = (const float*)d_in[2];   // (1024, 512)
    const float* b_hid = (const float*)d_in[3];   // (512)
    const float* W_emb = (const float*)d_in[4];   // (1024, 512)
    const float* b_emb = (const float*)d_in[5];   // (512)
    const float* att_v = (const float*)d_in[6];   // (512)

    float* ctx     = (float*)d_out;                       // (32, 1024)
    float* weights = (float*)d_out + BATCH * EMB_DIM;     // (32, 2048)

    char* ws = (char*)d_ws;
    float*    hb     = (float*)ws;                              //  64 KB
    float*    scores = (float*)(ws + 65536);                    // 256 KB
    _Float16* Wpk    = (_Float16*)(ws + 65536 + 262144);        //   1 MB

    k_hidproj <<<64,   256, 0, stream>>>(hid, W_hid, b_hid, b_emb, hb);
    k_wpack   <<<2048, 256, 0, stream>>>(W_emb, Wpk);
    k_attscore<<<2048, 128, 0, stream>>>(emb, Wpk, hb, att_v, scores);
    k_softmax <<<32,   256, 0, stream>>>(scores, weights);
    hipMemsetAsync(ctx, 0, BATCH * EMB_DIM * sizeof(float), stream);
    k_context <<<dim3(32, 4, 8), 256, 0, stream>>>(emb, weights, ctx);
}
